// EventSNNFlowNetLite_39668317946317
// MI455X (gfx1250) — matrix operand order repaired and checked by reading, not compiled, not run on hardware
//
#include <hip/hip_runtime.h>

// ---------------------------------------------------------------------------
// EventSNNFlowNetLite on MI455X (gfx1250).
// bf16 WMMA implicit-GEMM convs in NHWC with channel-innermost GEMM-K order:
// per 32-wide K-step the filter tap is constant and each lane's 16 B elements
// are 16 contiguous channels -> two global_load_b128 (no divisions, one
// bounds check). A-fragments are two b128 loads of pre-packed weight rows.
// One wave owns a 16-position strip and loops over all output-channel tiles,
// so each B fragment feeds MT WMMAs. LIF + 1x1 skip convs fused; fp32 state
// in workspace (~120MB, L2-resident). Flow head only at t = T-1.
// ---------------------------------------------------------------------------

typedef __bf16 bf16_t;
typedef __attribute__((ext_vector_type(16))) __bf16 v16bf;
typedef __attribute__((ext_vector_type(8)))  float  v8f;

namespace {

constexpr float kThr = 1.0f;
constexpr float kDecay = 0.5f;       // 1 - 1/TAU, TAU = 2
constexpr float kFlowScale = 16.0f;

constexpr int cdivc(int a, int b) { return (a + b - 1) / b; }
constexpr int rupc(int a, int b) { return cdivc(a, b) * b; }

union Frag16 {  // 16 bf16 = two 16B chunks
  uint4 u[2];
  v16bf v;
};
union F8 {      // 8 fp32 = two 16B chunks
  uint4 q[2];
  float f[8];
};
union H8 {      // 8 bf16 = one 16B chunk
  uint4 q;
  bf16_t h[8];
};

// -------------------------- elementwise kernels ----------------------------
__global__ __launch_bounds__(256) void f32_to_bf16_kernel(const float* __restrict__ in,
                                                          bf16_t* __restrict__ out, int n) {
  int i = blockIdx.x * 256 + threadIdx.x;
  if (i < n) out[i] = (bf16_t)in[i];
}

// w: [OC][IC][KH][KW] fp32 -> wp: [OCP][KP] bf16 zero-padded.
// cinner=1: packed k = rf*IC + c (tap-major, channel-innermost)
// cinner=0: packed k = c*KHW + rf (natural flattening)
__global__ __launch_bounds__(256) void pack_weights_kernel(const float* __restrict__ w,
                                                           bf16_t* __restrict__ wp,
                                                           int OC, int IC, int KHW,
                                                           int KP, int cinner, int total) {
  int i = blockIdx.x * 256 + threadIdx.x;
  if (i >= total) return;
  int m = i / KP, k = i % KP;
  const int K = IC * KHW;
  float v = 0.0f;
  if (m < OC && k < K) {
    int src;
    if (cinner) {
      int c = k % IC, rf = k / IC;
      src = (m * IC + c) * KHW + rf;
    } else {
      src = m * K + k;
    }
    v = w[src];
  }
  wp[i] = (bf16_t)v;
}

// me += (s - me)/tf ; also emit bf16 copy (skip-conv GEMM input)
__global__ __launch_bounds__(256) void mean_update_kernel(const bf16_t* __restrict__ s,
                                                          float* __restrict__ me,
                                                          bf16_t* __restrict__ me_bf,
                                                          float tf, int n) {
  int i = blockIdx.x * 256 + threadIdx.x;
  if (i >= n) return;
  float m = me[i];
  m += ((float)s[i] - m) / tf;
  me[i] = m;
  me_bf[i] = (bf16_t)m;
}

// ------------------- implicit-GEMM conv (+LIF, +skip) ----------------------
// Fragment layouts used here (wave32), half = lane>>4:
//  A 16x32 bf16: lane row M=lane&15 holds K = k0 + half*16 + {0..15} (contiguous).
//  B 32x16: lane col N=lane&15 holds K = k0 + half*16 + {0..15} (contiguous).
//  C/D 16x16 f32: VGPR r -> (M = r + half*8, N = lane&15).
// CINNER: activations NHWC, packed K = rf*IC + c, IC multiple of 32.
template <int NB, int IC, int OC, int KH, int KW, int STRIDE, int PAD,
          int IH, int IW, int BSTR, int OH, int OW,
          bool UP2, bool CINNER, bool DO_LIF, bool SKIP, int SKIP_C, bool FLOW_OUT>
__global__ __launch_bounds__(256) void conv_wmma_kernel(
    const bf16_t* __restrict__ act,      // CINNER: [b][IH][IW][IC], else NCHW
    const bf16_t* __restrict__ wp,       // [OCP][KP] packed bf16
    const float*  __restrict__ bias,     // [OC]
    const bf16_t* __restrict__ skip_act, // [NB][OH][OW][SKIP_C] NHWC
    const bf16_t* __restrict__ skip_wp,  // [OCP][rup(SKIP_C,32)]
    float*  __restrict__ mem,            // [NB][OH][OW][OC] LIF membrane (NHWC)
    bf16_t* __restrict__ out_bf,         // spike (+skip) output NHWC bf16
    float*  __restrict__ out_f32) {      // flow output NCHW fp32
  constexpr int K   = IC * KH * KW;
  constexpr int KP  = rupc(K, 32);
  constexpr int OCP = rupc(OC, 16);
  constexpr int MT  = OCP / 16;
  constexpr int NT  = (NB * OH * OW) / 16;   // OW is always a multiple of 16
  constexpr int IHE = UP2 ? IH * 2 : IH;
  constexpr int IWE = UP2 ? IW * 2 : IW;

  const int wid  = (int)((blockIdx.x * blockDim.x + threadIdx.x) >> 5);
  const int lane = (int)(threadIdx.x & 31);
  if (wid >= NT) return;                 // wave-uniform (grids are exact anyway)

  const int nl   = lane & 15;
  const int half = lane >> 4;
  const int n  = wid * 16 + nl;
  const int ow = n % OW;
  const int oh = (n / OW) % OH;
  const int b  = n / (OW * OH);
  const int ohs = oh * STRIDE - PAD;
  const int ows = ow * STRIDE - PAD;

  const bf16_t* arow  = wp + (size_t)nl * KP;   // per-lane weight row
  const int achunk = half * 16;

  v8f acc[MT] = {};
  for (int k0 = 0; k0 < KP; k0 += 32) {
    // ---- build B fragment once, reuse for all MT channel tiles ----
    v16bf bfrag = {};
    if constexpr (CINNER) {
      const int rf = k0 / IC;                  // filter tap, uniform
      const int fy = rf / KW;
      const int fx = rf % KW;
      const int ihf = ohs + fy;
      const int iwf = ows + fx;
      if (ihf >= 0 && ihf < IHE && iwf >= 0 && iwf < IWE) {
        const int ih = UP2 ? (ihf >> 1) : ihf;
        const int iw = UP2 ? (iwf >> 1) : iwf;
        const int cbase = (k0 % IC) + half * 16;  // 16 contiguous channels
        const bf16_t* bp = act + (size_t)b * BSTR + ((size_t)ih * IW + iw) * IC + cbase;
        Frag16 bb;
        bb.u[0] = *(const uint4*)(bp);
        bb.u[1] = *(const uint4*)(bp + 8);
        bfrag = bb.v;
      }
    } else {
#pragma unroll
      for (int j = 0; j < 16; ++j) {
        const int k = k0 + half * 16 + j;
        if (k < K) {
          const int c  = k / (KH * KW);
          const int r  = k % (KH * KW);
          const int fy = r / KW;
          const int fx = r % KW;
          const int ihf = ohs + fy;
          const int iwf = ows + fx;
          if (ihf >= 0 && ihf < IHE && iwf >= 0 && iwf < IWE)
            bfrag[j] = act[(size_t)b * BSTR + ((size_t)c * IH + ihf) * IW + iwf];
        }
      }
    }
#pragma unroll
    for (int mt = 0; mt < MT; ++mt) {
      Frag16 a;
      const bf16_t* rp = arow + (size_t)mt * 16 * KP + k0 + achunk;
      a.u[0] = *(const uint4*)(rp);
      a.u[1] = *(const uint4*)(rp + 8);
      acc[mt] = __builtin_amdgcn_wmma_f32_16x16x32_bf16(false, a.v, false, bfrag,
                                                        (short)0, acc[mt], false, false);
    }
  }

  v8f sacc[MT] = {};
  if constexpr (SKIP) {
    constexpr int KS = SKIP_C;                 // 32 or 64, multiple of 32
    const bf16_t* sp0 = skip_act + ((size_t)b * OH * OW + (size_t)oh * OW + ow) * KS;
    for (int k0 = 0; k0 < KS; k0 += 32) {
      Frag16 bb;
      const bf16_t* bp = sp0 + k0 + half * 16;
      bb.u[0] = *(const uint4*)(bp);
      bb.u[1] = *(const uint4*)(bp + 8);
#pragma unroll
      for (int mt = 0; mt < MT; ++mt) {
        Frag16 a;
        const bf16_t* rp = skip_wp + ((size_t)mt * 16 + nl) * KS + k0 + achunk;
        a.u[0] = *(const uint4*)(rp);
        a.u[1] = *(const uint4*)(rp + 8);
        sacc[mt] = __builtin_amdgcn_wmma_f32_16x16x32_bf16(false, a.v, false, bb.v,
                                                           (short)0, sacc[mt], false, false);
      }
    }
  }

  // ---- epilogue ----
  if constexpr (DO_LIF) {
    // NHWC: each lane's 8 accumulator rows are 8 consecutive channels.
#pragma unroll
    for (int mt = 0; mt < MT; ++mt) {
      const int mbase = mt * 16 + half * 8;
      const size_t obase = ((size_t)b * OH * OW + (size_t)oh * OW + ow) * OC + mbase;
      F8 bv, mv;
      bv.q[0] = *(const uint4*)(bias + mbase);
      bv.q[1] = *(const uint4*)(bias + mbase + 4);
      mv.q[0] = *(const uint4*)(mem + obase);
      mv.q[1] = *(const uint4*)(mem + obase + 4);
      H8 ov;
#pragma unroll
      for (int r = 0; r < 8; ++r) {
        const float v = acc[mt][r] + bv.f[r];
        float mm = mv.f[r] * kDecay + v;
        const float s = (mm - kThr) > 0.0f ? 1.0f : 0.0f;
        mm -= s * kThr;
        mv.f[r] = mm;
        float o = s;
        if constexpr (SKIP) o += sacc[mt][r];
        ov.h[r] = (bf16_t)o;
      }
      *(uint4*)(mem + obase) = mv.q[0];
      *(uint4*)(mem + obase + 4) = mv.q[1];
      *(uint4*)(out_bf + obase) = ov.q;
    }
  }
  if constexpr (FLOW_OUT) {
    // OC=2 -> scalar guarded scatter into NCHW d_out.
#pragma unroll
    for (int mt = 0; mt < MT; ++mt) {
#pragma unroll
      for (int r = 0; r < 8; ++r) {
        const int m = mt * 16 + r + half * 8;
        if (m < OC) {
          const float v = acc[mt][r] + bias[m];
          out_f32[(((size_t)b * OC + m) * OH + oh) * OW + ow] = v * kFlowScale;
        }
      }
    }
  }
}

}  // namespace

// ---------------------------------------------------------------------------
extern "C" void kernel_launch(void* const* d_in, const int* in_sizes, int n_in,
                              void* d_out, int out_size, void* d_ws, size_t ws_size,
                              hipStream_t stream) {
  (void)in_sizes; (void)n_in; (void)out_size; (void)ws_size;

  const float* x     = (const float*)d_in[0];
  const float* w_e1  = (const float*)d_in[1];
  const float* b_e1  = (const float*)d_in[2];
  const float* w_e2  = (const float*)d_in[3];
  const float* b_e2  = (const float*)d_in[4];
  const float* w_e3  = (const float*)d_in[5];
  const float* b_e3  = (const float*)d_in[6];
  const float* w_d3  = (const float*)d_in[7];
  const float* b_d3  = (const float*)d_in[8];
  const float* w_d2  = (const float*)d_in[9];
  const float* b_d2  = (const float*)d_in[10];
  const float* w_d1  = (const float*)d_in[11];
  const float* b_d1  = (const float*)d_in[12];
  const float* w_sk2 = (const float*)d_in[13];
  const float* w_sk1 = (const float*)d_in[14];
  const float* w_fl  = (const float*)d_in[15];
  const float* b_fl  = (const float*)d_in[16];

  constexpr int NB = 4, T = 8, H = 256, W = 256;
  const size_t S1  = (size_t)NB * 32 * 128 * 128;  // s1/me1/m1/md2/d2 elements
  const size_t S2  = (size_t)NB * 64 * 64 * 64;    // s2/me2/m2/md3/d3
  const size_t S3  = (size_t)NB * 128 * 32 * 32;   // s3/m3
  const size_t SD1 = (size_t)NB * 32 * 256 * 256;  // d1/md1
  const size_t XN  = (size_t)NB * T * 2 * H * W;   // x elements

  char* p = (char*)d_ws;
  auto alloc = [&p](size_t bytes) -> char* {
    char* r = p;
    p += (bytes + 255) & ~(size_t)255;
    return r;
  };

  // ---- zero-initialized state (single contiguous memset region) ----
  char* zbase = p;
  float*  m1   = (float*)alloc(S1 * 4);
  float*  m2   = (float*)alloc(S2 * 4);
  float*  m3   = (float*)alloc(S3 * 4);
  float*  md3  = (float*)alloc(S2 * 4);
  float*  md2  = (float*)alloc(S1 * 4);
  float*  md1  = (float*)alloc(SD1 * 4);
  float*  me1  = (float*)alloc(S1 * 4);
  float*  me2  = (float*)alloc(S2 * 4);
  bf16_t* me1b = (bf16_t*)alloc(S1 * 2);
  bf16_t* me2b = (bf16_t*)alloc(S2 * 2);
  const size_t zbytes = (size_t)(p - zbase);

  // ---- activations, NHWC (fully overwritten before read each step) ----
  bf16_t* xbf = (bf16_t*)alloc(XN * 2);   // NCHW copy of x (e1 generic path)
  bf16_t* s1  = (bf16_t*)alloc(S1 * 2);   // [4][128][128][32]
  bf16_t* s2  = (bf16_t*)alloc(S2 * 2);   // [4][64][64][64]
  bf16_t* s3  = (bf16_t*)alloc(S3 * 2);   // [4][32][32][128]
  bf16_t* d3s = (bf16_t*)alloc(S2 * 2);   // [4][64][64][64]
  bf16_t* d2s = (bf16_t*)alloc(S1 * 2);   // [4][128][128][32]
  bf16_t* d1s = (bf16_t*)alloc(SD1 * 2);  // [4][256][256][32]

  // ---- packed bf16 weights [rup(OC,16)][KP] ----
  bf16_t* wp_e1 = (bf16_t*)alloc((size_t)32 * 64 * 2);    // K=50 -> 64, natural
  bf16_t* wp_e2 = (bf16_t*)alloc((size_t)64 * 288 * 2);   // c-inner from here on
  bf16_t* wp_e3 = (bf16_t*)alloc((size_t)128 * 576 * 2);
  bf16_t* wp_d3 = (bf16_t*)alloc((size_t)64 * 1152 * 2);
  bf16_t* wp_d2 = (bf16_t*)alloc((size_t)32 * 576 * 2);
  bf16_t* wp_d1 = (bf16_t*)alloc((size_t)32 * 288 * 2);
  bf16_t* wp_s2 = (bf16_t*)alloc((size_t)64 * 64 * 2);
  bf16_t* wp_s1 = (bf16_t*)alloc((size_t)32 * 32 * 2);
  bf16_t* wp_fl = (bf16_t*)alloc((size_t)16 * 288 * 2);   // OC=2 -> padded 16

  hipMemsetAsync(zbase, 0, zbytes, stream);

  auto pack = [&](const float* w, bf16_t* wp, int OC, int IC, int KHW, int KP, int ci) {
    const int OCP = ((OC + 15) / 16) * 16;
    const int total = OCP * KP;
    pack_weights_kernel<<<(total + 255) / 256, 256, 0, stream>>>(w, wp, OC, IC, KHW, KP,
                                                                 ci, total);
  };
  pack(w_e1, wp_e1, 32, 2, 25, 64, 0);
  pack(w_e2, wp_e2, 64, 32, 9, 288, 1);
  pack(w_e3, wp_e3, 128, 64, 9, 576, 1);
  pack(w_d3, wp_d3, 64, 128, 9, 1152, 1);
  pack(w_d2, wp_d2, 32, 64, 9, 576, 1);
  pack(w_d1, wp_d1, 32, 32, 9, 288, 1);
  pack(w_sk2, wp_s2, 64, 64, 1, 64, 1);
  pack(w_sk1, wp_s1, 32, 32, 1, 32, 1);
  pack(w_fl, wp_fl, 2, 32, 9, 288, 1);

  f32_to_bf16_kernel<<<(int)((XN + 255) / 256), 256, 0, stream>>>(x, xbf, (int)XN);

  for (int t = 0; t < T; ++t) {
    const bf16_t* xt = xbf + (size_t)t * 2 * H * W;
    const float tf = (float)(t + 1);

    // e1: 2->32, 5x5 s2 p2, 256 -> 128 (generic NCHW gather, NHWC out)
    conv_wmma_kernel<NB, 2, 32, 5, 5, 2, 2, 256, 256, /*BSTR=*/T * 2 * H * W,
                     128, 128, false, false, true, false, 1, false>
        <<<512, 256, 0, stream>>>(xt, wp_e1, b_e1, nullptr, nullptr, m1, s1, nullptr);

    // e2: 32->64, 3x3 s2 p1, 128 -> 64
    conv_wmma_kernel<NB, 32, 64, 3, 3, 2, 1, 128, 128, 128 * 128 * 32,
                     64, 64, false, true, true, false, 1, false>
        <<<128, 256, 0, stream>>>(s1, wp_e2, b_e2, nullptr, nullptr, m2, s2, nullptr);

    // e3: 64->128, 3x3 s2 p1, 64 -> 32
    conv_wmma_kernel<NB, 64, 128, 3, 3, 2, 1, 64, 64, 64 * 64 * 64,
                     32, 32, false, true, true, false, 1, false>
        <<<32, 256, 0, stream>>>(s2, wp_e3, b_e3, nullptr, nullptr, m3, s3, nullptr);

    // running means (feed decoder skips this same step)
    mean_update_kernel<<<(int)((S1 + 255) / 256), 256, 0, stream>>>(s1, me1, me1b, tf, (int)S1);
    mean_update_kernel<<<(int)((S2 + 255) / 256), 256, 0, stream>>>(s2, me2, me2b, tf, (int)S2);

    // d3: up2(s3) 128->64, 3x3 s1 p1 @64, + skip2(me2) 1x1
    conv_wmma_kernel<NB, 128, 64, 3, 3, 1, 1, 32, 32, 32 * 32 * 128,
                     64, 64, true, true, true, true, 64, false>
        <<<128, 256, 0, stream>>>(s3, wp_d3, b_d3, me2b, wp_s2, md3, d3s, nullptr);

    // d2: up2(d3) 64->32, 3x3 @128, + skip1(me1) 1x1
    conv_wmma_kernel<NB, 64, 32, 3, 3, 1, 1, 64, 64, 64 * 64 * 64,
                     128, 128, true, true, true, true, 32, false>
        <<<512, 256, 0, stream>>>(d3s, wp_d2, b_d2, me1b, wp_s1, md2, d2s, nullptr);

    // d1: up2(d2) 32->32, 3x3 @256
    conv_wmma_kernel<NB, 32, 32, 3, 3, 1, 1, 128, 128, 128 * 128 * 32,
                     256, 256, true, true, true, false, 1, false>
        <<<2048, 256, 0, stream>>>(d2s, wp_d1, b_d1, nullptr, nullptr, md1, d1s, nullptr);

    // flow head only matters at the final step (output = flows[-1]); fm unused.
    if (t == T - 1) {
      conv_wmma_kernel<NB, 32, 2, 3, 3, 1, 1, 256, 256, 256 * 256 * 32,
                       256, 256, false, true, false, false, 1, true>
          <<<2048, 256, 0, stream>>>(d1s, wp_fl, b_fl, nullptr, nullptr, nullptr,
                                     nullptr, (float*)d_out);
    }
  }
}
